// get_loss_19061064860390
// MI455X (gfx1250) — compile-verified, matrix operand order checked by reading, not run on hardware
//
#include <hip/hip_runtime.h>
#include <math.h>

// ---------------------------------------------------------------------------
// WMMA tile helpers.
// A row m encodes (-2x, -2y, -2z, 1); B col n encodes (x, y, z, ||b||^2).
// D[m][n] = -2 a_m.b_n + ||b_n||^2  == d2(m,n) - ||a_m||^2.
// Since ||a_m||^2 is row-constant, mins/argmins are computed on D and the
// row norm is added back once after the column reduction.
// ---------------------------------------------------------------------------
typedef float v2f __attribute__((ext_vector_type(2)));
typedef float v8f __attribute__((ext_vector_type(8)));

#if defined(__AMDGCN__) && __has_builtin(__builtin_amdgcn_wmma_f32_16x16x4_f32)
typedef v2f areg_t;
__device__ __forceinline__ areg_t make_row_reg(float x, float y, float z, int half_) {
  // A 16x4 f32 layout: lanes 0-15 hold (k0,k1), lanes 16-31 hold (k2,k3) of row (lane&15)
  areg_t a;
  a.x = half_ ? (-2.0f * z) : (-2.0f * x);
  a.y = half_ ? 1.0f        : (-2.0f * y);
  return a;
}
__device__ __forceinline__ areg_t make_col_reg(float x, float y, float z, float bn2, int half_) {
  // B 4x16 f32, column-major mirror of A
  areg_t b;
  b.x = half_ ? z   : x;
  b.y = half_ ? bn2 : y;
  return b;
}
__device__ __forceinline__ v8f tile_mm(areg_t a, areg_t b) {
  v8f c = {};
  return __builtin_amdgcn_wmma_f32_16x16x4_f32(false, a, false, b, (short)0, c, false, false);
}
#else
// Fallback: f16 16x16x32 WMMA; K slots 0..3 used, rest zero.
typedef _Float16 v16h __attribute__((ext_vector_type(16)));
typedef v16h areg_t;
__device__ __forceinline__ areg_t make_row_reg(float x, float y, float z, int half_) {
  areg_t a = {};
  if (!half_) {
    a[0] = (_Float16)(-2.0f * x);
    a[1] = (_Float16)(-2.0f * y);
    a[2] = (_Float16)(-2.0f * z);
    a[3] = (_Float16)1.0f;
  }
  return a;
}
__device__ __forceinline__ areg_t make_col_reg(float x, float y, float z, float bn2, int half_) {
  areg_t b = {};
  if (!half_) {
    b[0] = (_Float16)x;
    b[1] = (_Float16)y;
    b[2] = (_Float16)z;
    b[3] = (_Float16)bn2;
  }
  return b;
}
__device__ __forceinline__ v8f tile_mm(areg_t a, areg_t b) {
  v8f c = {};
  return __builtin_amdgcn_wmma_f32_16x16x32_f16(false, a, false, b, (short)0, c, false, false);
}
#endif

#define E_DIR 0.57735027f

__device__ __forceinline__ void sample_point(const float* __restrict__ skel_xyz,
                                             const float* __restrict__ skel_rad,
                                             int base /* b*M */, int s,
                                             float& x, float& y, float& z) {
  int m = s >> 3, d = s & 7;
  const float* c = skel_xyz + (size_t)(base + m) * 3;
  float r = skel_rad[base + m];
  x = c[0] + r * ((d & 4) ? -E_DIR : E_DIR);
  y = c[1] + r * ((d & 2) ? -E_DIR : E_DIR);
  z = c[2] + r * ((d & 1) ? -E_DIR : E_DIR);
}

// ws: ws[0]=cd1 sum, ws[1]=cd2 sum, ws[2]=p2s sum, ws[3]=s2p sum
__global__ void k_zero(float* ws) {
  if (threadIdx.x < 8) ws[threadIdx.x] = 0.0f;
}

// ---------------------------------------------------------------------------
// cd1: rows = sample points (8M/batch), min over shape points. 2 row tiles/wave.
// ---------------------------------------------------------------------------
__global__ __launch_bounds__(32) void k_cd1(const float* __restrict__ skel_xyz,
                                            const float* __restrict__ skel_rad,
                                            const float* __restrict__ shape_xyz,
                                            float* __restrict__ ws, int M, int N) {
  const int lane = threadIdx.x, half_ = lane >> 4, r16 = lane & 15;
  const int tiles = (8 * M) / 32;
  const int b = blockIdx.x / tiles, t = blockIdx.x % tiles;

  float a0x, a0y, a0z, a1x, a1y, a1z;
  sample_point(skel_xyz, skel_rad, b * M, t * 32 + r16,      a0x, a0y, a0z);
  sample_point(skel_xyz, skel_rad, b * M, t * 32 + 16 + r16, a1x, a1y, a1z);
  float an2_0 = a0x * a0x + a0y * a0y + a0z * a0z;
  float an2_1 = a1x * a1x + a1y * a1y + a1z * a1z;
  areg_t areg0 = make_row_reg(a0x, a0y, a0z, half_);
  areg_t areg1 = make_row_reg(a1x, a1y, a1z, half_);

  float mn0[8], mn1[8];
#pragma unroll
  for (int i = 0; i < 8; ++i) { mn0[i] = 3.4e38f; mn1[i] = 3.4e38f; }

  const float* sb = shape_xyz + (size_t)b * N * 3;
  for (int ch = 0; ch < N / 16; ++ch) {
    int col = ch * 16 + r16;
    float bx = sb[col * 3 + 0], by = sb[col * 3 + 1], bz = sb[col * 3 + 2];
    float bn2 = bx * bx + by * by + bz * bz;
    areg_t breg = make_col_reg(bx, by, bz, bn2, half_);
    v8f c0 = tile_mm(areg0, breg);
    v8f c1 = tile_mm(areg1, breg);
#pragma unroll
    for (int i = 0; i < 8; ++i) { mn0[i] = fminf(mn0[i], c0[i]); mn1[i] = fminf(mn1[i], c1[i]); }
  }
#pragma unroll
  for (int m = 1; m < 16; m <<= 1)
#pragma unroll
    for (int i = 0; i < 8; ++i) {
      mn0[i] = fminf(mn0[i], __shfl_xor(mn0[i], m, 32));
      mn1[i] = fminf(mn1[i], __shfl_xor(mn1[i], m, 32));
    }
  // row norms: row i (+8*half_) lives in lane i+8*half_ (convergent shuffles)
  float rn0[8], rn1[8];
#pragma unroll
  for (int i = 0; i < 8; ++i) {
    rn0[i] = __shfl(an2_0, i + 8 * half_, 32);
    rn1[i] = __shfl(an2_1, i + 8 * half_, 32);
  }
  if (r16 == 0) {
    float s = 0.0f;
#pragma unroll
    for (int i = 0; i < 8; ++i) {
      s += sqrtf(fmaxf(mn0[i] + rn0[i], 1e-12f));
      s += sqrtf(fmaxf(mn1[i] + rn1[i], 1e-12f));
    }
    atomicAdd(&ws[0], s);
  }
}

// ---------------------------------------------------------------------------
// shape-rows: rows = shape points, 2 tiles/wave. Fuses cd2 (min over samples)
// and p2s (min+argmin over centers -> radius select).
// ---------------------------------------------------------------------------
__global__ __launch_bounds__(32) void k_shape(const float* __restrict__ skel_xyz,
                                              const float* __restrict__ skel_rad,
                                              const float* __restrict__ shape_xyz,
                                              float* __restrict__ ws, int M, int N) {
  const int lane = threadIdx.x, half_ = lane >> 4, r16 = lane & 15;
  const int tiles = N / 32;
  const int b = blockIdx.x / tiles, t = blockIdx.x % tiles;

  const float* sb = shape_xyz + (size_t)b * N * 3;
  int row0 = t * 32 + r16, row1 = row0 + 16;
  float a0x = sb[row0 * 3 + 0], a0y = sb[row0 * 3 + 1], a0z = sb[row0 * 3 + 2];
  float a1x = sb[row1 * 3 + 0], a1y = sb[row1 * 3 + 1], a1z = sb[row1 * 3 + 2];
  float an2_0 = a0x * a0x + a0y * a0y + a0z * a0z;
  float an2_1 = a1x * a1x + a1y * a1y + a1z * a1z;
  areg_t areg0 = make_row_reg(a0x, a0y, a0z, half_);
  areg_t areg1 = make_row_reg(a1x, a1y, a1z, half_);

  float mnA0[8], mnA1[8], mnB0[8], mnB1[8];
  int idx0[8], idx1[8];
#pragma unroll
  for (int i = 0; i < 8; ++i) {
    mnA0[i] = 3.4e38f; mnA1[i] = 3.4e38f;
    mnB0[i] = 3.4e38f; mnB1[i] = 3.4e38f;
    idx0[i] = 0; idx1[i] = 0;
  }

  // cd2: columns = sample points
  for (int ch = 0; ch < (8 * M) / 16; ++ch) {
    int col = ch * 16 + r16;
    float bx, by, bz;
    sample_point(skel_xyz, skel_rad, b * M, col, bx, by, bz);
    float bn2 = bx * bx + by * by + bz * bz;
    areg_t breg = make_col_reg(bx, by, bz, bn2, half_);
    v8f c0 = tile_mm(areg0, breg);
    v8f c1 = tile_mm(areg1, breg);
#pragma unroll
    for (int i = 0; i < 8; ++i) { mnA0[i] = fminf(mnA0[i], c0[i]); mnA1[i] = fminf(mnA1[i], c1[i]); }
  }

  // p2s: columns = sphere centers, track argmin for radius select
  for (int ch = 0; ch < M / 16; ++ch) {
    int col = ch * 16 + r16;
    const float* cc = skel_xyz + (size_t)(b * M + col) * 3;
    float bx = cc[0], by = cc[1], bz = cc[2];
    float bn2 = bx * bx + by * by + bz * bz;
    areg_t breg = make_col_reg(bx, by, bz, bn2, half_);
    v8f c0 = tile_mm(areg0, breg);
    v8f c1 = tile_mm(areg1, breg);
#pragma unroll
    for (int i = 0; i < 8; ++i) {
      bool l0 = c0[i] < mnB0[i];
      mnB0[i] = l0 ? c0[i] : mnB0[i];
      idx0[i] = l0 ? col : idx0[i];
      bool l1 = c1[i] < mnB1[i];
      mnB1[i] = l1 ? c1[i] : mnB1[i];
      idx1[i] = l1 ? col : idx1[i];
    }
  }

#pragma unroll
  for (int m = 1; m < 16; m <<= 1) {
#pragma unroll
    for (int i = 0; i < 8; ++i) {
      mnA0[i] = fminf(mnA0[i], __shfl_xor(mnA0[i], m, 32));
      mnA1[i] = fminf(mnA1[i], __shfl_xor(mnA1[i], m, 32));
      float od0 = __shfl_xor(mnB0[i], m, 32);
      int   oi0 = __shfl_xor(idx0[i], m, 32);
      bool  l0 = od0 < mnB0[i];
      mnB0[i] = l0 ? od0 : mnB0[i];
      idx0[i] = l0 ? oi0 : idx0[i];
      float od1 = __shfl_xor(mnB1[i], m, 32);
      int   oi1 = __shfl_xor(idx1[i], m, 32);
      bool  l1 = od1 < mnB1[i];
      mnB1[i] = l1 ? od1 : mnB1[i];
      idx1[i] = l1 ? oi1 : idx1[i];
    }
  }
  float rn0[8], rn1[8];
#pragma unroll
  for (int i = 0; i < 8; ++i) {
    rn0[i] = __shfl(an2_0, i + 8 * half_, 32);
    rn1[i] = __shfl(an2_1, i + 8 * half_, 32);
  }

  if (r16 == 0) {
    float sA = 0.0f, sB = 0.0f;
#pragma unroll
    for (int i = 0; i < 8; ++i) {
      sA += sqrtf(fmaxf(mnA0[i] + rn0[i], 1e-12f));
      sA += sqrtf(fmaxf(mnA1[i] + rn1[i], 1e-12f));
      float md0 = sqrtf(fmaxf(mnB0[i] + rn0[i], 1e-12f));
      float d0 = md0 - skel_rad[b * M + idx0[i]];
      float md1 = sqrtf(fmaxf(mnB1[i] + rn1[i], 1e-12f));
      float d1 = md1 - skel_rad[b * M + idx1[i]];
      sB += d0 * d0 + d1 * d1;
    }
    atomicAdd(&ws[1], sA);
    atomicAdd(&ws[2], sB);
  }
}

// ---------------------------------------------------------------------------
// s2p: rows = sphere centers (M/batch), min over shape points, (min_d - r)^2
// ---------------------------------------------------------------------------
__global__ __launch_bounds__(32) void k_s2p(const float* __restrict__ skel_xyz,
                                            const float* __restrict__ skel_rad,
                                            const float* __restrict__ shape_xyz,
                                            float* __restrict__ ws, int M, int N) {
  const int lane = threadIdx.x, half_ = lane >> 4, r16 = lane & 15;
  const int tiles = M / 16;
  const int b = blockIdx.x / tiles, t = blockIdx.x % tiles;

  const float* cc = skel_xyz + (size_t)(b * M + t * 16 + r16) * 3;
  float ax = cc[0], ay = cc[1], az = cc[2];
  float an2 = ax * ax + ay * ay + az * az;
  areg_t areg = make_row_reg(ax, ay, az, half_);

  float mn[8];
#pragma unroll
  for (int i = 0; i < 8; ++i) mn[i] = 3.4e38f;

  const float* sb = shape_xyz + (size_t)b * N * 3;
  for (int ch = 0; ch < N / 16; ++ch) {
    int col = ch * 16 + r16;
    float bx = sb[col * 3 + 0], by = sb[col * 3 + 1], bz = sb[col * 3 + 2];
    float bn2 = bx * bx + by * by + bz * bz;
    areg_t breg = make_col_reg(bx, by, bz, bn2, half_);
    v8f c = tile_mm(areg, breg);
#pragma unroll
    for (int i = 0; i < 8; ++i) mn[i] = fminf(mn[i], c[i]);
  }
#pragma unroll
  for (int m = 1; m < 16; m <<= 1)
#pragma unroll
    for (int i = 0; i < 8; ++i) mn[i] = fminf(mn[i], __shfl_xor(mn[i], m, 32));
  float rn[8];
#pragma unroll
  for (int i = 0; i < 8; ++i) rn[i] = __shfl(an2, i + 8 * half_, 32);

  if (r16 == 0) {
    float s = 0.0f;
#pragma unroll
    for (int i = 0; i < 8; ++i) {
      int rowM = t * 16 + i + half_ * 8;
      float md = sqrtf(fmaxf(mn[i] + rn[i], 1e-12f));
      float d = md - skel_rad[b * M + rowM];
      s += d * d;
    }
    atomicAdd(&ws[3], s);
  }
}

// Final: radius sum reduction + weighted combine into the scalar output.
__global__ void k_final(const float* __restrict__ skel_rad, const float* __restrict__ ws,
                        const float* __restrict__ w1p, const float* __restrict__ w2p,
                        float* __restrict__ out, int BM, int M, int N) {
  __shared__ float sm[256];
  float s = 0.0f;
  for (int i = threadIdx.x; i < BM; i += blockDim.x) s += skel_rad[i];
  sm[threadIdx.x] = s;
  __syncthreads();
  for (int off = 128; off > 0; off >>= 1) {
    if ((int)threadIdx.x < off) sm[threadIdx.x] += sm[threadIdx.x + off];
    __syncthreads();
  }
  if (threadIdx.x == 0) {
    float radsum = sm[0];
    float cd1 = ws[0] / (8.0f * (float)M);
    float cd2 = ws[1] / (float)N;
    float p2s = ws[2] / (float)N;
    float s2p = ws[3] / (float)M;
    float w1 = *w1p, w2 = *w2p;
    out[0] = (cd1 + cd2) + w1 * (p2s + s2p) + w2 * (-radsum / (float)M);
  }
}

extern "C" void kernel_launch(void* const* d_in, const int* in_sizes, int n_in,
                              void* d_out, int out_size, void* d_ws, size_t ws_size,
                              hipStream_t stream) {
  const float* skel_xyz  = (const float*)d_in[0];  // (B,M,3)
  const float* skel_rad  = (const float*)d_in[1];  // (B,M,1)
  const float* shape_xyz = (const float*)d_in[3];  // (B,N,3)
  const float* w1        = (const float*)d_in[5];
  const float* w2        = (const float*)d_in[6];
  float* ws  = (float*)d_ws;
  float* out = (float*)d_out;

  // derive dims: in_sizes[1]=B*M, in_sizes[4]=B*M*M, in_sizes[3]=B*N*3
  const int M = in_sizes[4] / in_sizes[1];
  const int B = in_sizes[1] / M;
  const int N = in_sizes[3] / (3 * B);

  k_zero<<<1, 32, 0, stream>>>(ws);
  k_cd1<<<B * (8 * M / 32), 32, 0, stream>>>(skel_xyz, skel_rad, shape_xyz, ws, M, N);
  k_shape<<<B * (N / 32), 32, 0, stream>>>(skel_xyz, skel_rad, shape_xyz, ws, M, N);
  k_s2p<<<B * (M / 16), 32, 0, stream>>>(skel_xyz, skel_rad, shape_xyz, ws, M, N);
  k_final<<<1, 256, 0, stream>>>(skel_rad, ws, w1, w2, out, B * M, M, N);
}
